// QuantizationLayer_29119878267455
// MI455X (gfx1250) — matrix operand order repaired and checked by reading, not compiled, not run on hardware
//
#include <hip/hip_runtime.h>

typedef __attribute__((ext_vector_type(16))) _Float16 v16h;
typedef __attribute__((ext_vector_type(2)))  _Float16 h2;
typedef __attribute__((ext_vector_type(8)))  float    v8f;

#define CC 9
#define HH 260
#define WW 346
#define BBATCH 16
#define EE (16 * 32768)
#define HID 100
#define NTILE 7              // ceil(100/16) N tiles
#define KTILE 4              // ceil(100/32) K steps
#define PLANE (WW * HH)      // 89960
#define NVOX (2 * CC * HH * WW * BBATCH)  // 25908480
#define REDSTRIDE 20         // padded row stride (floats): 20*r mod 64 distinct -> no bank conflicts

__global__ void zero_out_kernel(float4* __restrict__ out, int n4) {
  int i = blockIdx.x * blockDim.x + threadIdx.x;
  int stride = gridDim.x * blockDim.x;
  for (; i < n4; i += stride) out[i] = make_float4(0.f, 0.f, 0.f, 0.f);
}

__global__ __launch_bounds__(256) void est_voxel_kernel(
    const int* __restrict__ xs, const int* __restrict__ ys,
    const float* __restrict__ ts, const int* __restrict__ ps,
    const int* __restrict__ bs,
    const float* __restrict__ W1, const float* __restrict__ b1,
    const float* __restrict__ W2, const float* __restrict__ b2,
    const float* __restrict__ W3, const float* __restrict__ b3,
    float* __restrict__ out)
{
  // B operands (W2^T tiles) in 16-bit B VGPR layout:
  // operand o = nb*KTILE+kb; slot = lane; element i -> K = kb*32 + (lane>=16?8:0) + (i<8 ? i : i+8)
  __shared__ __attribute__((aligned(32))) _Float16 sW2[NTILE * KTILE * 512];
  // (W1,b1) as f16 pairs pre-swizzled to the A layout. Slot = kb*32+lane, 16 h2 per slot:
  // [0..7] = W1 pairs, [8..15] = b1 pairs; pair j covers k0(j) = kb*32 + (j<4 ? 2j : 2j+8) + hi*8
  __shared__ __attribute__((aligned(16))) _Float16 sWBh[KTILE * 32 * 16 * 2];
  __shared__ float sB2[NTILE * 16];
  __shared__ float sW3[NTILE * 16];
  // per-wave transpose-reduce scratch: 16 rows x REDSTRIDE floats
  __shared__ __attribute__((aligned(16))) float sRed[8 * 16 * REDSTRIDE];

  const int tid = threadIdx.x;

  for (int s = tid; s < KTILE * 32 * 16; s += 256) {
    int j2 = s & 15, slot = s >> 4;
    int laneslot = slot & 31, kb = slot >> 5;
    int hi = laneslot >> 4;
    int j = j2 & 7;
    int k0 = kb * 32 + ((j < 4) ? 2 * j : 2 * j + 8) + hi * 8;
    const float* src = (j2 < 8) ? W1 : b1;
    h2 v;
    v.x = (k0 < HID) ? (_Float16)src[k0] : (_Float16)0.f;
    v.y = (k0 + 1 < HID) ? (_Float16)src[k0 + 1] : (_Float16)0.f;
    *reinterpret_cast<h2*>(&sWBh[2 * s]) = v;
  }
  for (int s = tid; s < NTILE * KTILE * 512; s += 256) {
    int i = s & 15, laneslot = (s >> 4) & 31, o = s >> 9;
    int nb = o / KTILE, kb = o % KTILE;
    int n = nb * 16 + (laneslot & 15);
    int k = kb * 32 + ((laneslot >= 16) ? 8 : 0) + ((i < 8) ? i : i + 8);
    float v = (n < HID && k < HID) ? W2[n * HID + k] : 0.f;
    sW2[s] = (_Float16)v;
  }
  for (int s = tid; s < NTILE * 16; s += 256) {
    sB2[s] = (s < HID) ? b2[s] : 0.f;
    sW3[s] = (s < HID) ? W3[s] : 0.f;
  }
  __syncthreads();

  const int lane = tid & 31;
  const int m = lane & 15;
  const int hi = lane >> 4;
  const int widx = tid >> 5;
  const int gwid = blockIdx.x * 8 + widx;
  const int nwaves = gridDim.x * 8;
  const float b3v = b3[0];
  float* const red = &sRed[widx * 16 * REDSTRIDE];

  // hoist (W1,b1) f16 pairs into registers: invariant across tiles and bins
  h2 rw[KTILE][8], rb[KTILE][8];
  #pragma unroll
  for (int kb = 0; kb < KTILE; ++kb) {
    const h2* p = reinterpret_cast<const h2*>(&sWBh[(kb * 32 + lane) * 32]);
    #pragma unroll
    for (int j = 0; j < 8; ++j) { rw[kb][j] = p[j]; rb[kb][j] = p[8 + j]; }
  }

  const int ntiles = EE / 16;
  for (int tile = gwid; tile < ntiles; tile += nwaves) {
    const int e0 = tile * 16;
    const float tloc = ts[e0 + m];     // t of the A row this lane feeds

    // scatter metadata: lane r (< 16) owns event r of the tile
    int sidx = 0;
    float stv = 0.f;
    if (lane < 16) {
      int e = e0 + lane;
      sidx = xs[e] + WW * ys[e] + (PLANE * CC) * ps[e] + (2 * PLANE * CC) * bs[e];
      stv = ts[e];
    }

    #pragma unroll 1
    for (int bin = 0; bin < CC; ++bin) {
      const _Float16 th = (_Float16)(tloc - 0.125f * (float)bin);
      const h2 t2 = {th, th};

      // ---- layer 1 in packed f16, directly in A-operand layout (no DS, no cvt)
      union { v16h v; h2 p[8]; } A[KTILE];
      #pragma unroll
      for (int kb = 0; kb < KTILE; ++kb) {
        #pragma unroll
        for (int j = 0; j < 8; ++j) {
          h2 x = __builtin_elementwise_fma(t2, rw[kb][j], rb[kb][j]);
          h2 xs_ = x * (h2){(_Float16)0.1f, (_Float16)0.1f};
          A[kb].p[j] = __builtin_elementwise_max(x, xs_);   // leaky relu
        }
      }

      // ---- layer 2 (WMMA, dual accumulators) + fused layer 3 epilogue
      float partial[8];
      #pragma unroll
      for (int r = 0; r < 8; ++r) partial[r] = 0.f;

      #pragma unroll 1
      for (int nb = 0; nb < NTILE; ++nb) {
        const _Float16* base = &sW2[nb * KTILE * 512 + lane * 16];
        v16h b0 = *reinterpret_cast<const v16h*>(base);
        v16h b1m = *reinterpret_cast<const v16h*>(base + 512);
        v16h b2m = *reinterpret_cast<const v16h*>(base + 1024);
        v16h b3m = *reinterpret_cast<const v16h*>(base + 1536);
        v8f c0 = {}, c1 = {};
        c0 = __builtin_amdgcn_wmma_f32_16x16x32_f16(false, A[0].v, false, b0,  (short)0, c0, false, false);
        c1 = __builtin_amdgcn_wmma_f32_16x16x32_f16(false, A[1].v, false, b1m, (short)0, c1, false, false);
        c0 = __builtin_amdgcn_wmma_f32_16x16x32_f16(false, A[2].v, false, b2m, (short)0, c0, false, false);
        c1 = __builtin_amdgcn_wmma_f32_16x16x32_f16(false, A[3].v, false, b3m, (short)0, c1, false, false);
        // C layout: lane column n = lane&15, VGPR r -> row (hi*8 + r)
        const int n = nb * 16 + m;
        const float b2n = sB2[n];
        const float w3n = sW3[n];          // zero-padded -> padded N contributes 0
        #pragma unroll
        for (int r = 0; r < 8; ++r) {
          float x = (c0[r] + c1[r]) + b2n;
          x = fmaxf(x, 0.1f * x);
          partial[r] = fmaf(x, w3n, partial[r]);
        }
      }

      // ---- transpose-reduce via per-wave LDS scratch (bank-conflict-free)
      #pragma unroll
      for (int r = 0; r < 8; ++r)
        red[(hi * 8 + r) * REDSTRIDE + m] = partial[r];

      const float4* rrow = reinterpret_cast<const float4*>(&red[m * REDSTRIDE]);
      float4 s0 = rrow[0], s1 = rrow[1], s2 = rrow[2], s3 = rrow[3];
      float sum = ((s0.x + s0.y) + (s0.z + s0.w)) + ((s1.x + s1.y) + (s1.z + s1.w))
                + ((s2.x + s2.y) + (s2.z + s2.w)) + ((s3.x + s3.y) + (s3.z + s3.w));

      if (lane < 16) {
        int idx = sidx + PLANE * bin;
        idx = min(max(idx, 0), NVOX - 1);
        unsafeAtomicAdd(&out[idx], stv * (sum + b3v));
      }
    }
  }
}

extern "C" void kernel_launch(void* const* d_in, const int* in_sizes, int n_in,
                              void* d_out, int out_size, void* d_ws, size_t ws_size,
                              hipStream_t stream) {
  const int*   xs = (const int*)d_in[0];
  const int*   ys = (const int*)d_in[1];
  const float* ts = (const float*)d_in[2];
  const int*   ps = (const int*)d_in[3];
  const int*   bs = (const int*)d_in[4];
  const float* W1 = (const float*)d_in[5];
  const float* b1 = (const float*)d_in[6];
  const float* W2 = (const float*)d_in[7];
  const float* b2 = (const float*)d_in[8];
  const float* W3 = (const float*)d_in[9];
  const float* b3 = (const float*)d_in[10];
  float* out = (float*)d_out;

  zero_out_kernel<<<2048, 256, 0, stream>>>((float4*)out, NVOX / 4);
  est_voxel_kernel<<<1024, 256, 0, stream>>>(xs, ys, ts, ps, bs,
                                             W1, b1, W2, b2, W3, b3, out);
}
